// LstmModel_20401094656567
// MI455X (gfx1250) — matrix operand order repaired and checked by reading, not compiled, not run on hardware
//
#include <hip/hip_runtime.h>
#include <hip/hip_bf16.h>

typedef __attribute__((ext_vector_type(16))) _Float16 v16h;
typedef __attribute__((ext_vector_type(8)))  _Float16 v8h;
typedef __attribute__((ext_vector_type(4)))  _Float16 v4h;
typedef __attribute__((ext_vector_type(8)))  float    v8f;

#define LSTM_T   8192
#define LSTM_H   1024
#define LSTM_D   64
#define G4H      4096          // 4*H
#define OUT_T    1024
#define OUT_D    64
#define REC_NB   64            // blocks in recurrence (co-resident, << #WGPs)
#define REC_HB   (LSTM_H / REC_NB)   // 16 h-elements per block

// ---------------------------------------------------------------------------
// fp32 -> fp16 elementwise convert (grid-stride)
// ---------------------------------------------------------------------------
__global__ void f32_to_f16_kernel(const float* __restrict__ src,
                                  _Float16* __restrict__ dst, int n) {
  int i = blockIdx.x * blockDim.x + threadIdx.x;
  int stride = gridDim.x * blockDim.x;
  for (; i < n; i += stride) dst[i] = (_Float16)src[i];
}

__global__ void bias_sum_kernel(const float* __restrict__ a,
                                const float* __restrict__ b,
                                float* __restrict__ out, int n) {
  int i = blockIdx.x * blockDim.x + threadIdx.x;
  if (i < n) out[i] = a[i] + b[i];
}

// ---------------------------------------------------------------------------
// WMMA GEMM: C[M,N] = A[M,K] * B[N,K]^T (+ bias[N]), fp16 in, fp32 acc.
// Block = 256 threads = 8 waves arranged 4(M) x 2(N); wave tile 32x32 =
// 2x2 wmma accumulators (4 x v_wmma_f32_16x16x32_f16 per 32-K step, 2x
// operand reuse on both A and B fragments). Block tile 128x64.
// Requires: M % 128 == 0, N % 64 == 0, K % 32 == 0.
// ---------------------------------------------------------------------------
__device__ __forceinline__ v16h load_a_frag(const _Float16* p) {
  // 16-bit A 16x32 layout: per lane two 16B chunks [kb,kb+8) and [kb+16,kb+24)
  v8h lo = *(const v8h*)(p);
  v8h hi = *(const v8h*)(p + 16);
  return __builtin_shufflevector(lo, hi, 0,1,2,3,4,5,6,7,8,9,10,11,12,13,14,15);
}

__global__ __launch_bounds__(256)
void gemm_wmma_kernel(const _Float16* __restrict__ A, int lda,
                      const _Float16* __restrict__ B, int ldb,
                      const float* __restrict__ bias,
                      float* __restrict__ Cf,
                      _Float16* __restrict__ Ch, int ldc,
                      int K, int relu) {
  const int lane = threadIdx.x & 31;
  const int wave = threadIdx.x >> 5;              // 0..7
  const int m0 = blockIdx.y * 128 + (wave >> 1) * 32;
  const int n0 = blockIdx.x * 64  + (wave & 1) * 32;

  // A fragments: lane&15 = M row, lane>>4 selects K sub-chunk {0,8}.
  const int kg = lane >> 4;
  const int am = m0 + (lane & 15);
  const _Float16* Ap0 = A + (size_t)am * lda + kg * 8;
  const _Float16* Ap1 = Ap0 + (size_t)16 * lda;

  // B fragments (B[k][n] = W[n][k]): lane&15 = N col, lane>>4 selects K sub-chunk {0,8}.
  const int bn = n0 + (lane & 15);
  const _Float16* Bp0 = B + (size_t)bn * ldb + kg * 8;
  const _Float16* Bp1 = Bp0 + (size_t)16 * ldb;

  v8f c00 = {}, c01 = {}, c10 = {}, c11 = {};
  for (int k0 = 0; k0 < K; k0 += 32) {
    if ((k0 & 63) == 0) {           // one near-cache prefetch per 128B line
      __builtin_prefetch(Ap0 + k0 + 256, 0, 3);
      __builtin_prefetch(Ap1 + k0 + 256, 0, 3);
      __builtin_prefetch(Bp0 + k0 + 256, 0, 3);
      __builtin_prefetch(Bp1 + k0 + 256, 0, 3);
    }
    v16h a0 = load_a_frag(Ap0 + k0);
    v16h a1 = load_a_frag(Ap1 + k0);
    v16h b0 = load_a_frag(Bp0 + k0);
    v16h b1 = load_a_frag(Bp1 + k0);
    c00 = __builtin_amdgcn_wmma_f32_16x16x32_f16(false, a0, false, b0,
                                                 (short)0, c00, false, false);
    c01 = __builtin_amdgcn_wmma_f32_16x16x32_f16(false, a0, false, b1,
                                                 (short)0, c01, false, false);
    c10 = __builtin_amdgcn_wmma_f32_16x16x32_f16(false, a1, false, b0,
                                                 (short)0, c10, false, false);
    c11 = __builtin_amdgcn_wmma_f32_16x16x32_f16(false, a1, false, b1,
                                                 (short)0, c11, false, false);
  }

  // C layout: VGPR r -> M = rowbase + r + 8*(lane>=16), N = colbase + (lane&15)
  const int rsub = (lane >> 4) * 8;
  const int csub = lane & 15;
  const v8f* accs[4] = { &c00, &c01, &c10, &c11 };
  #pragma unroll
  for (int tile = 0; tile < 4; ++tile) {
    const int rowbase = m0 + (tile >> 1) * 16 + rsub;
    const int col     = n0 + (tile & 1) * 16 + csub;
    const v8f& acc = *accs[tile];
    float bv = bias ? bias[col] : 0.f;
    #pragma unroll
    for (int r = 0; r < 8; ++r) {
      int row = rowbase + r;
      float v = acc[r] + bv;
      if (relu) v = fmaxf(v, 0.f);
      if (Cf) Cf[(size_t)row * ldc + col] = v;
      if (Ch) Ch[(size_t)row * ldc + col] = (_Float16)v;
    }
  }
}

// ---------------------------------------------------------------------------
// Persistent LSTM recurrence. 64 blocks x 256 threads. Each block owns 16
// h-elements (=> 64 gate rows of W_hh, 128 KB fp16, staged once into LDS).
// Thread map: e = tid>>4 (local h elem), g = (tid>>2)&3 (gate), ch = tid&3
// (256-col K chunk). Dot over LDS weights, butterfly-reduce over ch, shfl-
// gather the 4 gates; all 16 threads of a group replicate (c,h); writer lane
// stores h (fp16) into the layer output Y[T,H], which is also next step's
// h input. Device-wide step barrier via atomic counter + s_sleep spin.
// ---------------------------------------------------------------------------
__global__ __launch_bounds__(256)
void lstm_rec_kernel(const _Float16* __restrict__ Whh,  // [4096,1024] fp16
                     const _Float16* __restrict__ pre,  // [T,4096] fp16
                     _Float16* __restrict__ Y,          // [T,1024] fp16
                     unsigned int* __restrict__ cnt, int T) {
  extern __shared__ unsigned char smem[];
  _Float16* wlds = (_Float16*)smem;                                  // 64*1024
  _Float16* hrow = (_Float16*)(smem + 64 * 1024 * sizeof(_Float16)); // 1024

  const int tid  = threadIdx.x;
  const int lane = tid & 31;
  const int e  = tid >> 4;
  const int g  = (tid >> 2) & 3;
  const int ch = tid & 3;
  const int j    = blockIdx.x * REC_HB + e;   // global h index owned
  const int grow = g * LSTM_H + j;            // gate row in [0,4096)

  // Stage this block's 64 W_hh rows into LDS: row lr=(e*4+g) at wlds[lr*1024].
  for (int idx = tid; idx < 64 * 128; idx += 256) {   // 128 x v8h per row
    int lr = idx >> 7;
    int kc = (idx & 127) * 8;
    int le = lr >> 2, lg = lr & 3;
    int gr = lg * LSTM_H + blockIdx.x * REC_HB + le;
    *(v8h*)(wlds + lr * 1024 + kc) = *(const v8h*)(Whh + (size_t)gr * LSTM_H + kc);
  }

  float c = 0.f;
  const v8h* wp = (const v8h*)(wlds + (e * 4 + g) * 1024 + ch * 256);

  for (int t = 0; t < T; ++t) {
    // Broadcast h_{t-1} (2 KB) into LDS.
    {
      int off = tid * 4;
      v4h hv = {};
      if (t > 0) hv = *(const v4h*)(Y + (size_t)(t - 1) * LSTM_H + off);
      *(v4h*)(hrow + off) = hv;
    }
    __syncthreads();
    __builtin_prefetch(pre + (size_t)(t + 1) * G4H + grow, 0, 3);

    // 256-wide fp16 dot with fp32 accumulation (LDS b128 reads).
    const v8h* hp = (const v8h*)(hrow + ch * 256);
    float s = 0.f;
    #pragma unroll 8
    for (int k = 0; k < 32; ++k) {
      v8h w = wp[k], h8 = hp[k];
      #pragma unroll
      for (int u = 0; u < 8; ++u) s = fmaf((float)w[u], (float)h8[u], s);
    }
    s += __shfl_xor(s, 1);   // butterfly: all 4 ch lanes hold full sum
    s += __shfl_xor(s, 2);
    float gate = s + (float)pre[(size_t)t * G4H + grow];

    int base = lane & 16;    // e-group base within the wave
    float iv = __shfl(gate, base + 0, 32);
    float fv = __shfl(gate, base + 4, 32);
    float gv = __shfl(gate, base + 8, 32);
    float ov = __shfl(gate, base + 12, 32);

    float is = 1.f / (1.f + __expf(-iv));
    float fs = 1.f / (1.f + __expf(-fv));
    float gt = tanhf(gv);
    float os = 1.f / (1.f + __expf(-ov));
    c = fs * c + is * gt;
    float h = os * tanhf(c);
    if ((tid & 15) == 0) Y[(size_t)t * LSTM_H + j] = (_Float16)h;

    // Device-wide step barrier.
    __threadfence();
    __syncthreads();
    if (tid == 0) {
      __hip_atomic_fetch_add(cnt, 1u, __ATOMIC_RELEASE, __HIP_MEMORY_SCOPE_AGENT);
      unsigned target = (unsigned)REC_NB * (unsigned)(t + 1);
      while (__hip_atomic_load(cnt, __ATOMIC_ACQUIRE, __HIP_MEMORY_SCOPE_AGENT) < target)
        __builtin_amdgcn_s_sleep(1);
    }
    __syncthreads();
    __threadfence();
  }
}

// ---------------------------------------------------------------------------
// Host launcher
// ---------------------------------------------------------------------------
extern "C" void kernel_launch(void* const* d_in, const int* in_sizes, int n_in,
                              void* d_out, int out_size, void* d_ws, size_t ws_size,
                              hipStream_t stream) {
  (void)in_sizes; (void)n_in; (void)out_size; (void)ws_size;
  const float* x     = (const float*)d_in[0];
  const float* Wih0  = (const float*)d_in[1];
  const float* Whh0  = (const float*)d_in[2];
  const float* bih0  = (const float*)d_in[3];
  const float* bhh0  = (const float*)d_in[4];
  const float* Wih1  = (const float*)d_in[5];
  const float* Whh1  = (const float*)d_in[6];
  const float* bih1  = (const float*)d_in[7];
  const float* bhh1  = (const float*)d_in[8];
  const float* W2    = (const float*)d_in[9];
  const float* b2    = (const float*)d_in[10];
  const float* W3    = (const float*)d_in[11];
  const float* b3    = (const float*)d_in[12];

  // Workspace carve-up (256B aligned slabs).
  unsigned char* ws = (unsigned char*)d_ws;
  size_t off = 0;
  auto carve = [&](size_t bytes) -> void* {
    void* p = ws + off;
    off = (off + bytes + 255) & ~(size_t)255;
    return p;
  };
  _Float16* x16    = (_Float16*)carve((size_t)LSTM_T * LSTM_D * 2);
  _Float16* Wih0h  = (_Float16*)carve((size_t)G4H * LSTM_D * 2);
  _Float16* Whh0h  = (_Float16*)carve((size_t)G4H * LSTM_H * 2);
  _Float16* Wih1h  = (_Float16*)carve((size_t)G4H * LSTM_H * 2);
  _Float16* Whh1h  = (_Float16*)carve((size_t)G4H * LSTM_H * 2);
  _Float16* W2h    = (_Float16*)carve((size_t)LSTM_H * LSTM_H * 2);
  _Float16* W3h    = (_Float16*)carve((size_t)OUT_D * LSTM_H * 2);
  float*    b0s    = (float*)carve((size_t)G4H * 4);
  float*    b1s    = (float*)carve((size_t)G4H * 4);
  _Float16* pre16  = (_Float16*)carve((size_t)LSTM_T * G4H * 2);   // 64 MB
  _Float16* Y1     = (_Float16*)carve((size_t)LSTM_T * LSTM_H * 2);
  _Float16* Y2     = (_Float16*)carve((size_t)LSTM_T * LSTM_H * 2);
  _Float16* Y4h    = (_Float16*)carve((size_t)OUT_T * LSTM_H * 2);
  unsigned* cnts   = (unsigned*)carve(256);

  hipMemsetAsync(cnts, 0, 256, stream);   // capture-legal barrier-counter init

  auto conv = [&](const float* s, _Float16* d, int n) {
    int blocks = (n + 255) / 256; if (blocks > 1024) blocks = 1024;
    f32_to_f16_kernel<<<blocks, 256, 0, stream>>>(s, d, n);
  };
  conv(x,    x16,   LSTM_T * LSTM_D);
  conv(Wih0, Wih0h, G4H * LSTM_D);
  conv(Whh0, Whh0h, G4H * LSTM_H);
  conv(Wih1, Wih1h, G4H * LSTM_H);
  conv(Whh1, Whh1h, G4H * LSTM_H);
  conv(W2,   W2h,   LSTM_H * LSTM_H);
  conv(W3,   W3h,   OUT_D * LSTM_H);
  bias_sum_kernel<<<(G4H + 255) / 256, 256, 0, stream>>>(bih0, bhh0, b0s, G4H);
  bias_sum_kernel<<<(G4H + 255) / 256, 256, 0, stream>>>(bih1, bhh1, b1s, G4H);

  const size_t rec_lds = 64 * 1024 * sizeof(_Float16) + LSTM_H * sizeof(_Float16);

  // Layer 0: preact = x @ Wih0^T + (b_ih0+b_hh0)  [8192 x 4096], K=64
  gemm_wmma_kernel<<<dim3(G4H / 64, LSTM_T / 128), 256, 0, stream>>>(
      x16, LSTM_D, Wih0h, LSTM_D, b0s, nullptr, pre16, G4H, LSTM_D, 0);
  lstm_rec_kernel<<<REC_NB, 256, rec_lds, stream>>>(Whh0h, pre16, Y1, &cnts[0], LSTM_T);

  // Layer 1: preact = Y1 @ Wih1^T + (b_ih1+b_hh1)  [8192 x 4096], K=1024
  gemm_wmma_kernel<<<dim3(G4H / 64, LSTM_T / 128), 256, 0, stream>>>(
      Y1, LSTM_H, Wih1h, LSTM_H, b1s, nullptr, pre16, G4H, LSTM_H, 0);
  lstm_rec_kernel<<<REC_NB, 256, rec_lds, stream>>>(Whh1h, pre16, Y2, &cnts[32], LSTM_T);

  // Tail: y4 = relu(Y2[last 1024] @ W2^T + b2) -> fp16
  const _Float16* y2tail = Y2 + (size_t)(LSTM_T - OUT_T) * LSTM_H;
  gemm_wmma_kernel<<<dim3(LSTM_H / 64, OUT_T / 128), 256, 0, stream>>>(
      y2tail, LSTM_H, W2h, LSTM_H, b2, nullptr, Y4h, LSTM_H, LSTM_H, 1);

  // y5 = y4 @ W3^T + b3 -> fp32 d_out [1024 x 64]
  gemm_wmma_kernel<<<dim3(OUT_D / 64, OUT_T / 128), 256, 0, stream>>>(
      Y4h, LSTM_H, W3h, LSTM_H, b3, (float*)d_out, nullptr, OUT_D, LSTM_H, 0);
}
